// QuadEnhancer_47957604827151
// MI455X (gfx1250) — matrix operand order repaired and checked by reading, not compiled, not run on hardware
//
#include <hip/hip_runtime.h>

typedef __attribute__((ext_vector_type(16))) _Float16 v16h;
typedef __attribute__((ext_vector_type(8)))  float    v8f;
typedef __attribute__((ext_vector_type(4)))  float    f4;

namespace {
constexpr int D      = 4096;           // feature dim
constexpr int KW     = 16;             // band width
constexpr int NCB    = D / 16;         // 256 column blocks
constexpr int NSTRIP = 8192 / 16;      // 512 row strips
constexpr int NCHUNK = 8;              // column chunks per strip
constexpr int CPW    = NCB / NCHUNK;   // 32 tiles per wave
}

// Build the 256 banded weight matrices (16x32 f16 each) directly in the
// per-lane A-operand layout this code uses (contiguous k order, same as B):
//   lane L: M = L%16 (=c), g = L/16; half slot j -> K = 16g+j
//   A[c,K] = w[K-c-1, 16C+c] if 1 <= K-c <= 16 else 0
__global__ void __launch_bounds__(256)
qe_prep_weights(const float* __restrict__ w, _Float16* __restrict__ wsa) {
  int idx = blockIdx.x * 256 + threadIdx.x;   // [0, NCB*512)
  int C = idx >> 9;
  int r = idx & 511;
  int L = r >> 4;            // lane 0..31
  int j = r & 15;            // half slot within lane
  int c = L & 15;            // A-matrix row (M)
  int g = L >> 4;
  int k = 16 * g + j;
  int i = k - c;
  float val = 0.0f;
  if (i >= 1 && i <= KW) val = w[(i - 1) * D + (C * 16 + c)];
  wsa[idx] = (_Float16)val;
}

// One wave computes a sequence of 16x16 output tiles along one 16-row strip.
// S^T(16x16) = A(16x32 banded weights, f16) x B(32x16 x-slice, f16), f32 acc.
// out = x * (1 + S) done on the D-matrix layout (contiguous per-lane columns).
__global__ void __launch_bounds__(256)
qe_main(const float* __restrict__ x, const _Float16* __restrict__ wsa,
        float* __restrict__ out) {
  const int lane  = threadIdx.x & 31;
  const int wid   = blockIdx.x * 8 + (threadIdx.x >> 5);
  const int strip = wid >> 3;            // 0..511
  const int chunk = wid & 7;             // 0..7
  const int g     = lane >> 4;
  const int n     = lane & 15;           // N index: row within strip
  const long rowOff = (long)(strip * 16 + n) * D;
  const float* xrow = x + rowOff;
  float*       orow = out + rowOff;

  const int C0 = chunk * CPW;
  for (int C = C0; C < C0 + CPW; ++C) {
    const int cb = C * 16;

    // ---- B operand: lane holds x[row][(cb + 16g + j) mod D], j = 0..15
    const int bc = cb + 16 * g;
    f4 xf[4];
#pragma unroll
    for (int t = 0; t < 4; ++t)
      xf[t] = *(const f4*)(xrow + ((bc + 4 * t) & (D - 1)));  // runs never straddle wrap
    v16h bm;
#pragma unroll
    for (int t = 0; t < 4; ++t) {
      bm[4 * t + 0] = (_Float16)xf[t][0];
      bm[4 * t + 1] = (_Float16)xf[t][1];
      bm[4 * t + 2] = (_Float16)xf[t][2];
      bm[4 * t + 3] = (_Float16)xf[t][3];
    }

    // ---- A operand: precomputed banded weights, already in A layout (L2 resident)
    v16h am = *(const v16h*)(wsa + (size_t)C * 512 + lane * 16);

    // ---- S^T tile = A x B  (one V_WMMA_F32_16X16X32_F16)
    v8f acc = {};
    acc = __builtin_amdgcn_wmma_f32_16x16x32_f16(
        /*neg_a=*/false, am, /*neg_b=*/false, bm,
        /*c_mod=*/(short)0, acc, /*reuse_a=*/false, /*reuse_b=*/false);

    // ---- out[row][cb + 8g + v] = x * (1 + S), v = 0..7 (contiguous columns)
    const float* xc = xrow + cb + 8 * g;
    f4 xlo = *(const f4*)xc;
    f4 xhi = *(const f4*)(xc + 4);
    f4 olo, ohi;
#pragma unroll
    for (int v = 0; v < 4; ++v) {
      olo[v] = fmaf(xlo[v], acc[v],     xlo[v]);
      ohi[v] = fmaf(xhi[v], acc[v + 4], xhi[v]);
    }
    float* op = orow + cb + 8 * g;
    *(f4*)op       = olo;
    *(f4*)(op + 4) = ohi;
  }
}

extern "C" void kernel_launch(void* const* d_in, const int* in_sizes, int n_in,
                              void* d_out, int out_size, void* d_ws, size_t ws_size,
                              hipStream_t stream) {
  (void)in_sizes; (void)n_in; (void)out_size; (void)ws_size;
  const float* x = (const float*)d_in[0];
  const float* w = (const float*)d_in[1];
  float* out     = (float*)d_out;
  _Float16* wsa  = (_Float16*)d_ws;   // needs 256 KB

  // Build banded weight matrices in WMMA A layout, then run the fused kernel.
  qe_prep_weights<<<(NCB * 512) / 256, 256, 0, stream>>>(w, wsa);
  qe_main<<<(NSTRIP * NCHUNK) / 8, 256, 0, stream>>>(x, wsa, out);
}
